// GraphOperation_4045859192917
// MI455X (gfx1250) — matrix operand order repaired and checked by reading, not compiled, not run on hardware
//
#include <hip/hip_runtime.h>

// ---------------------------------------------------------------------------
// Fused band-GCN for MI455X (gfx1250, wave32, WMMA).
// out[b,c,n] = sum_{j in [n-9,n+9]\{n}} dinv[j]*dinv[n]*h[b,j,c]
//            + dinv[n]^2 * h[b,n,c] + bias[c]
// where h = xs @ Wg and xs[b,n,k] = x[b, n&255, 16*k + (n>>8)].
// GEMM done as bf16-split (hi/lo) on v_wmma_f32_16x16x32_bf16.
// ---------------------------------------------------------------------------

typedef __attribute__((ext_vector_type(16))) __bf16 v16bf;
typedef __attribute__((ext_vector_type(8)))  __bf16 v8bf;
typedef __attribute__((ext_vector_type(8)))  float  v8f;

#define N_NODES  4096
#define C_DIM    256
#define ROWS_OUT 32          // output rows per workgroup
#define MROWS    64          // h rows computed: [R0-16, R0+48) covers +-9 halo
#define APAD     264         // bf16 row stride for A tiles (conflict-free b128)
#define HPAD     257         // f32 row stride for h tile (stride%64 == 1)
#define A_ELEMS  (MROWS * APAD)                  // 16896 bf16 per matrix
#define SMEM_BYTES (2 * A_ELEMS * 2)             // 67584 B >= 64*257*4 h tile

__launch_bounds__(256, 1)
__global__ void band_gcn_wmma_kernel(const float* __restrict__ x,
                                     const float* __restrict__ Wg,
                                     const float* __restrict__ bias,
                                     float* __restrict__ out)
{
    __shared__ __align__(16) unsigned char smem_raw[SMEM_BYTES];
    __bf16* Ahi = (__bf16*)smem_raw;             // [MROWS][APAD]
    __bf16* Alo = Ahi + A_ELEMS;                 // [MROWS][APAD]
    float*  Hs  = (float*)smem_raw;              // [MROWS][HPAD] (reuses A region)

    const int t    = threadIdx.x;                // 256 threads = 8 waves
    const int lane = t & 31;
    const int wave = t >> 5;
    const int blk  = blockIdx.x;
    const int b    = blk >> 7;                   // 128 row-blocks per batch
    const int R0   = (blk & 127) * ROWS_OUT;

    const float* xb = x + (size_t)b * C_DIM * N_NODES;

    // ---- Phase 0: permuted gather -> bf16 hi/lo split into LDS -------------
    // xs[n][k] = x[b, n&255, 16*k + (n>>8)]; thread t supplies k = t.
    for (int row = 0; row < MROWS; ++row) {
        const int n = R0 - 16 + row;
        float v = 0.0f;
        if ((unsigned)n < (unsigned)N_NODES) {
            const int lo = n & 255, hi = n >> 8;
            v = xb[(size_t)lo * N_NODES + t * 16 + hi];
        }
        const __bf16 vh = (__bf16)v;
        const __bf16 vl = (__bf16)(v - (float)vh);
        Ahi[row * APAD + t] = vh;
        Alo[row * APAD + t] = vl;
    }
    __syncthreads();

    // ---- Phase 1: h = xs @ Wg via bf16x3 WMMA ------------------------------
    const int koff = (lane < 16) ? 0 : 8;        // B frag: per-lane K offset
    const int aoff = (lane < 16) ? 0 : 8;        // A frag: per-lane K offset
    const int arow = lane & 15;                  // A frag: M row within tile
    const int cloc = lane & 15;                  // B/D frag: N col within tile

    v8f acc[2][4];
    #pragma unroll
    for (int ci = 0; ci < 2; ++ci)
        #pragma unroll
        for (int rt = 0; rt < 4; ++rt)
            acc[ci][rt] = (v8f)(0.0f);

    #pragma unroll
    for (int ci = 0; ci < 2; ++ci) {
        const int ct = wave + 8 * ci;            // this wave's column tiles
        const int c  = ct * 16 + cloc;
        #pragma unroll
        for (int kb = 0; kb < 8; ++kb) {
            const int k0 = kb * 32;
            // B fragment: 16 K values of Wg column c (lane-strided, coalesced
            // across lanes; Wg is 256 KB and stays L2-resident).
            v16bf bhi, blo;
            #pragma unroll
            for (int e = 0; e < 16; ++e) {
                const float v = Wg[(k0 + koff + e + (e >= 8 ? 8 : 0)) * C_DIM + c];
                const __bf16 vh = (__bf16)v;
                bhi[e] = vh;
                blo[e] = (__bf16)(v - (float)vh);
            }
            #pragma unroll
            for (int rt = 0; rt < 4; ++rt) {
                const __bf16* ah = &Ahi[(rt * 16 + arow) * APAD + k0 + aoff];
                const __bf16* al = &Alo[(rt * 16 + arow) * APAD + k0 + aoff];
                const v8bf ah0 = *(const v8bf*)ah;
                const v8bf ah1 = *(const v8bf*)(ah + 16);
                const v8bf al0 = *(const v8bf*)al;
                const v8bf al1 = *(const v8bf*)(al + 16);
                v16bf afh, afl;
                #pragma unroll
                for (int e = 0; e < 8; ++e) {
                    afh[e] = ah0[e]; afh[e + 8] = ah1[e];
                    afl[e] = al0[e]; afl[e + 8] = al1[e];
                }
                // fp32-ish product: Ahi*Bhi + Ahi*Blo + Alo*Bhi
                acc[ci][rt] = __builtin_amdgcn_wmma_f32_16x16x32_bf16(
                    false, afh, false, bhi, (short)0, acc[ci][rt], false, false);
                acc[ci][rt] = __builtin_amdgcn_wmma_f32_16x16x32_bf16(
                    false, afh, false, blo, (short)0, acc[ci][rt], false, false);
                acc[ci][rt] = __builtin_amdgcn_wmma_f32_16x16x32_bf16(
                    false, afl, false, bhi, (short)0, acc[ci][rt], false, false);
            }
        }
    }
    __syncthreads();   // everyone done reading A before h overwrites it

    // ---- Phase 2: spill accumulators to LDS h tile -------------------------
    // 16x16 f32 C/D layout: VGPR v, lanes 0-15 -> M=v, lanes 16-31 -> M=v+8.
    const int rofs = (lane < 16) ? 0 : 8;
    #pragma unroll
    for (int ci = 0; ci < 2; ++ci) {
        const int ct = wave + 8 * ci;
        #pragma unroll
        for (int rt = 0; rt < 4; ++rt) {
            #pragma unroll
            for (int v = 0; v < 8; ++v) {
                Hs[(rt * 16 + v + rofs) * HPAD + ct * 16 + cloc] = acc[ci][rt][v];
            }
        }
    }
    __syncthreads();

    // ---- Phase 3: banded aggregation + self-loop + bias, transposed store --
    const int n  = R0 + (t & 31);                // lane-consecutive n
    const int lb = 16 + (t & 31);                // local h row of n
    const int c0 = (t >> 5) * 32;                // wave-uniform column base

    const float dn = rsqrtf((float)(min(N_NODES, n + 10) - max(0, n - 9)));
    float wts[19];
    #pragma unroll
    for (int o = 0; o < 19; ++o) {
        const int j = n + o - 9;
        if (o == 9) {
            wts[o] = dn * dn;                    // self-loop: 1/deg(n)
        } else if ((unsigned)j < (unsigned)N_NODES) {
            const float dj = rsqrtf((float)(min(N_NODES, j + 10) - max(0, j - 9)));
            wts[o] = dj * dn;
        } else {
            wts[o] = 0.0f;
        }
    }

    for (int cc = 0; cc < 32; ++cc) {
        const int c = c0 + cc;
        float s = 0.0f;
        #pragma unroll
        for (int o = 0; o < 19; ++o)
            s += wts[o] * Hs[(lb + o - 9) * HPAD + c];   // stride 257: no conflicts
        out[((size_t)b * C_DIM + c) * N_NODES + n] = s + bias[c];  // coalesced in n
    }
}

extern "C" void kernel_launch(void* const* d_in, const int* in_sizes, int n_in,
                              void* d_out, int out_size, void* d_ws, size_t ws_size,
                              hipStream_t stream)
{
    const float* x    = (const float*)d_in[0];   // (8, 256, 64, 64) f32
    const float* Wg   = (const float*)d_in[1];   // (256, 256) f32
    const float* bias = (const float*)d_in[2];   // (256,) f32
    float* out = (float*)d_out;                  // (8, 256, 64, 64) f32

    const dim3 grid(8 * (N_NODES / ROWS_OUT));   // 1024 workgroups
    band_gcn_wmma_kernel<<<grid, 256, 0, stream>>>(x, Wg, bias, out);
}
